// RibosomeCompressor_37709812859432
// MI455X (gfx1250) — matrix operand order repaired and checked by reading, not compiled, not run on hardware
//
#include <hip/hip_runtime.h>
#include <hip/hip_bf16.h>

// ---------------------------------------------------------------------------
// RibosomeCompressor for MI455X (gfx1250, wave32, WMMA)
// K/V projections folded through the 16-query cross-attention (exact after
// softmax).  f16 operands -> WMMA fragments as contiguous b128 loads; heads
// fused so token_embeds is streamed once per consumer kernel; async
// global->LDS staging (ASYNCcnt) in the attn@X kernel.
// ---------------------------------------------------------------------------

constexpr int B_  = 8;
constexpr int S_  = 4096;
constexpr int H_  = 1024;
constexpr int NC_ = 16;
constexpr int NH_ = 4;
constexpr int HD_ = 256;

typedef __attribute__((ext_vector_type(16))) _Float16 v16h;
typedef __attribute__((ext_vector_type(8)))  _Float16 v8h;
typedef __attribute__((ext_vector_type(8)))  float    v8f;
typedef __attribute__((ext_vector_type(4)))  int      v4i_;

#if defined(__has_builtin)
#if __has_builtin(__builtin_amdgcn_global_load_async_to_lds_b128) && \
    __has_builtin(__builtin_amdgcn_s_wait_asynccnt)
#define ASYNC_LDS 1
#endif
#endif

#ifdef ASYNC_LDS
typedef __attribute__((address_space(1))) v4i_ as1_v4i;   // global int4
typedef __attribute__((address_space(3))) v4i_ as3_v4i;   // LDS int4
#endif

__device__ inline v8f wmma16x16x32(v16h a, v16h b, v8f c) {
  return __builtin_amdgcn_wmma_f32_16x16x32_f16(
      false, a, false, b, (short)0, c, false, false);
}

// A fragment (16x32 f16 tile, row-major, row stride lda in halves).
// Per lane (half=lane>>4, m=lane&15): elems 0..15 = K[16h..16h+15]
// contiguous  -> two contiguous 128-bit loads.
__device__ inline v16h load_a_frag_h(const _Float16* A, int lda, int lane) {
  const int half = lane >> 4, m = lane & 15;
  const _Float16* p = A + (long)m * lda + 16 * half;
  v8h lo = *(const v8h*)p;
  v8h hi = *(const v8h*)(p + 8);
  return __builtin_shufflevector(lo, hi, 0,1,2,3,4,5,6,7,8,9,10,11,12,13,14,15);
}

// B fragment for X @ W^T (B[k][n] = W[n][k]); W row-major f16, stride ldw.
// Per lane (n=lane&15, half=lane>>4): elems = K[16h..16h+15] contiguous.
__device__ inline v16h load_bT_frag_h(const _Float16* W, int ldw, int lane) {
  const int half = lane >> 4, n = lane & 15;
  const _Float16* p = W + (long)n * ldw + 16 * half;
  v8h lo = *(const v8h*)p;
  v8h hi = *(const v8h*)(p + 8);
  return __builtin_shufflevector(lo, hi, 0,1,2,3,4,5,6,7,8,9,10,11,12,13,14,15);
}

// --------------------------------------------------------------------------
// f32 -> f16 bulk convert, 8 elements/thread
// --------------------------------------------------------------------------
__global__ void k_cvt16(const float* __restrict__ src, _Float16* __restrict__ dst) {
  const long i = ((long)blockIdx.x * blockDim.x + threadIdx.x) * 8;
  const float4 a = *(const float4*)(src + i);
  const float4 b = *(const float4*)(src + i + 4);
  v8h h;
  h[0] = (_Float16)a.x; h[1] = (_Float16)a.y; h[2] = (_Float16)a.z; h[3] = (_Float16)a.w;
  h[4] = (_Float16)b.x; h[5] = (_Float16)b.y; h[6] = (_Float16)b.z; h[7] = (_Float16)b.w;
  *(v8h*)(dst + i) = h;
}

// --------------------------------------------------------------------------
// Kernel 1: importance = sigmoid(gelu(x@W1^T+b1)@W2^T+b2)
// --------------------------------------------------------------------------
__global__ void k_importance(const _Float16* __restrict__ xh,
                             const _Float16* __restrict__ W1h,
                             const float* __restrict__ b1,
                             const float* __restrict__ W2,
                             const float* __restrict__ b2,
                             float* __restrict__ imp) {
  const int wave = threadIdx.x >> 5, lane = threadIdx.x & 31;
  const int half = lane >> 4, nn = lane & 15;
  const long t0 = (long)blockIdx.x * 128 + wave * 16;
  const _Float16* xrow = xh + t0 * H_;
  float partial[8];
#pragma unroll
  for (int r = 0; r < 8; ++r) partial[r] = 0.0f;

  for (int ng = 0; ng < 4; ++ng) {
    v8f acc[4] = {};
    for (int kt = 0; kt < H_ / 32; ++kt) {
      __builtin_prefetch((const void*)(xrow + (kt + 4) * 32), 0, 3);
      v16h a = load_a_frag_h(xrow + kt * 32, H_, lane);
#pragma unroll
      for (int t = 0; t < 4; ++t) {
        v16h b = load_bT_frag_h(W1h + (long)((ng * 4 + t) * 16) * H_ + kt * 32, H_, lane);
        acc[t] = wmma16x16x32(a, b, acc[t]);
      }
    }
#pragma unroll
    for (int t = 0; t < 4; ++t) {
      const int nt = ng * 4 + t;
      const float b1n = b1[nt * 16 + nn];
      const float w2n = W2[nt * 16 + nn];
#pragma unroll
      for (int r = 0; r < 8; ++r) {
        const float v = acc[t][r] + b1n;
        const float g = 0.5f * v * (1.0f + erff(v * 0.70710678118f));
        partial[r] += g * w2n;
      }
    }
  }
#pragma unroll
  for (int r = 0; r < 8; ++r) {
    float v = partial[r];
    for (int off = 1; off < 16; off <<= 1) v += __shfl_xor(v, off, 32);
    partial[r] = v;
  }
  if (nn == 0) {
#pragma unroll
    for (int r = 0; r < 8; ++r) {
      const float z = partial[r] + b2[0];
      imp[t0 + half * 8 + r] = 1.0f / (1.0f + __expf(-z));
    }
  }
}

// --------------------------------------------------------------------------
// Kernel 2a/2b: tiny q projection + folded K projection
// --------------------------------------------------------------------------
__global__ void k_qproj(const float* __restrict__ cq, const float* __restrict__ Wq,
                        const float* __restrict__ bq, float* __restrict__ q) {
  const int idx = blockIdx.x * blockDim.x + threadIdx.x;
  const int qi = idx >> 10, o = idx & (H_ - 1);
  const float* a = cq + qi * H_;
  const float* w = Wq + (long)o * H_;
  float s = bq[o];
  for (int j = 0; j < H_; ++j) s += a[j] * w[j];
  q[idx] = s;
}

__global__ void k_qkfold(const float* __restrict__ q, const float* __restrict__ Wk,
                         float* __restrict__ qk) {
  const int idx = blockIdx.x * blockDim.x + threadIdx.x;
  const int j = idx & (H_ - 1);
  const int qi = (idx >> 10) & (NC_ - 1);
  const int h = idx >> 14;
  float s = 0.0f;
  for (int d = 0; d < HD_; ++d)
    s += q[qi * H_ + h * HD_ + d] * Wk[(long)(h * HD_ + d) * H_ + j];
  qk[idx] = s;
}

// --------------------------------------------------------------------------
// Kernel 3: scores[b,h,qi,s] = imp[b,s]*(qk[h,qi,:].x[b,s,:])/16
// x tile is the A operand (m = s) so x is streamed ONCE; 4 head accumulators
// share it; per lane the 8 acc elements are 8 consecutive s -> b128 stores.
// --------------------------------------------------------------------------
__global__ void k_scores(const _Float16* __restrict__ xh, const _Float16* __restrict__ qkh,
                         const float* __restrict__ imp, float* __restrict__ scores) {
  const int wave = threadIdx.x >> 5, lane = threadIdx.x & 31;
  const int half = lane >> 4, qi = lane & 15;
  const int b = blockIdx.y;
  const int s0 = blockIdx.x * 128 + wave * 16;
  const _Float16* X = xh + ((long)b * S_ + s0) * H_;
  v8f acc[NH_] = {};
  for (int kt = 0; kt < H_ / 32; ++kt) {
    __builtin_prefetch((const void*)(X + (kt + 4) * 32), 0, 3);
    v16h a = load_a_frag_h(X + kt * 32, H_, lane);
#pragma unroll
    for (int h = 0; h < NH_; ++h) {
      v16h bf = load_bT_frag_h(qkh + (long)(h * NC_) * H_ + kt * 32, H_, lane);
      acc[h] = wmma16x16x32(a, bf, acc[h]);
    }
  }
  const float4 i0 = *(const float4*)(imp + (long)b * S_ + s0 + 8 * half);
  const float4 i1 = *(const float4*)(imp + (long)b * S_ + s0 + 8 * half + 4);
#pragma unroll
  for (int h = 0; h < NH_; ++h) {
    float* row = scores + (((long)b * NH_ + h) * NC_ + qi) * S_ + s0 + 8 * half;
    float4 o0, o1;
    o0.x = acc[h][0] * 0.0625f * i0.x;  o0.y = acc[h][1] * 0.0625f * i0.y;
    o0.z = acc[h][2] * 0.0625f * i0.z;  o0.w = acc[h][3] * 0.0625f * i0.w;
    o1.x = acc[h][4] * 0.0625f * i1.x;  o1.y = acc[h][5] * 0.0625f * i1.y;
    o1.z = acc[h][6] * 0.0625f * i1.z;  o1.w = acc[h][7] * 0.0625f * i1.w;
    *(float4*)row = o0;
    *(float4*)(row + 4) = o1;
  }
}

// --------------------------------------------------------------------------
// Kernel 4: row softmax over S (in place) + f16 copy for the WMMA consumer
// --------------------------------------------------------------------------
__global__ void k_softmax(float* __restrict__ attn, _Float16* __restrict__ attnh) {
  __shared__ float red[256];
  const int tid = threadIdx.x;
  float* p = attn + (long)blockIdx.x * S_;
  _Float16* ph = attnh + (long)blockIdx.x * S_;
  float lmax = -1e30f;
  for (int i = tid; i < S_; i += 256) lmax = fmaxf(lmax, p[i]);
  red[tid] = lmax; __syncthreads();
  for (int s = 128; s > 0; s >>= 1) { if (tid < s) red[tid] = fmaxf(red[tid], red[tid + s]); __syncthreads(); }
  const float m = red[0]; __syncthreads();
  float lsum = 0.0f;
  for (int i = tid; i < S_; i += 256) { const float e = __expf(p[i] - m); p[i] = e; lsum += e; }
  red[tid] = lsum; __syncthreads();
  for (int s = 128; s > 0; s >>= 1) { if (tid < s) red[tid] += red[tid + s]; __syncthreads(); }
  const float inv = 1.0f / red[0];
  for (int i = tid; i < S_; i += 256) {
    const float v = p[i] * inv;
    p[i] = v;
    ph[i] = (_Float16)v;
  }
}

// --------------------------------------------------------------------------
// Kernel 5: attn_weights[b,qi,s] = mean_h attn[b,h,qi,s]
// --------------------------------------------------------------------------
__global__ void k_attnmean(const float* __restrict__ attn, float* __restrict__ aw) {
  const long idx = (long)blockIdx.x * blockDim.x + threadIdx.x;
  const int s = (int)(idx & (S_ - 1));
  const int qi = (int)((idx >> 12) & (NC_ - 1));
  const int b = (int)(idx >> 16);
  float acc = 0.0f;
#pragma unroll
  for (int h = 0; h < NH_; ++h)
    acc += attn[(((long)b * NH_ + h) * NC_ + qi) * S_ + s];
  aw[idx] = 0.25f * acc;
}

// --------------------------------------------------------------------------
// Kernel 6: ctx_pre[b,h,qi,j] = sum_s attn[b,h,qi,s]*x[b,s,j], all 4 heads
// fused so each staged LDS x-tile feeds 4 WMMAs per wave per K-step.
// Staging uses GLOBAL_LOAD_ASYNC_TO_LDS_B128 (ASYNCcnt) when available.
// --------------------------------------------------------------------------
__global__ void k_ctxpre(const _Float16* __restrict__ attnh, const _Float16* __restrict__ xh,
                         _Float16* __restrict__ ctxpre_h) {
  __shared__ _Float16 tile[2][32 * 128];       // 32 K x 128 N f16, 2 buffers
  const int tid = threadIdx.x;
  const int lane = tid & 31;
  const int wave = tid >> 5;
  const int half = lane >> 4, n = lane & 15;
  const int b = blockIdx.y;
  const int n0 = blockIdx.x * 128;
  const int nw = wave * 16;
  const _Float16* Xb = xh + (long)b * S_ * H_ + n0;
  const int srow = tid >> 3, scol = (tid & 7) * 16;   // 8 threads per K-row

  auto stage = [&](int buf, int kt) {
    const _Float16* src = Xb + ((long)kt * 32 + srow) * H_ + scol;
    _Float16* dst = &tile[buf][srow * 128 + scol];
#ifdef ASYNC_LDS
    __builtin_amdgcn_global_load_async_to_lds_b128((as1_v4i*)src, (as3_v4i*)dst, 0, 0);
    __builtin_amdgcn_global_load_async_to_lds_b128((as1_v4i*)(src + 8), (as3_v4i*)(dst + 8), 0, 0);
#else
    *(v8h*)dst       = *(const v8h*)src;
    *(v8h*)(dst + 8) = *(const v8h*)(src + 8);
#endif
  };
  auto stage_wait = [&]() {
#ifdef ASYNC_LDS
    __builtin_amdgcn_s_wait_asynccnt(0);
#endif
  };

  stage(0, 0);
  stage_wait();
  __syncthreads();

  v8f acc[NH_] = {};
  for (int kt = 0; kt < S_ / 32; ++kt) {
    const int cur = kt & 1;
    if (kt + 1 < S_ / 32) stage(cur ^ 1, kt + 1);
    // B fragment from LDS (shared across the 4 head WMMAs)
    v16h bf;
    const _Float16* T = &tile[cur][0];
#pragma unroll
    for (int j = 0; j < 8; ++j) {
      const int k = 16 * half + 2 * j;
      bf[2 * j]     = T[k * 128 + nw + n];
      bf[2 * j + 1] = T[(k + 1) * 128 + nw + n];
    }
#pragma unroll
    for (int h = 0; h < NH_; ++h) {
      v16h a = load_a_frag_h(attnh + (((long)b * NH_ + h) * NC_) * S_ + kt * 32, S_, lane);
      acc[h] = wmma16x16x32(a, bf, acc[h]);
    }
    stage_wait();
    __syncthreads();
  }
#pragma unroll
  for (int h = 0; h < NH_; ++h) {
#pragma unroll
    for (int r = 0; r < 8; ++r) {
      const int qi = r + 8 * half;
      ctxpre_h[(((long)b * NH_ + h) * NC_ + qi) * H_ + n0 + nw + n] = (_Float16)acc[h][r];
    }
  }
}

// --------------------------------------------------------------------------
// Kernel 7: ctx[b,qi,o] = ctx_pre[b,o/HD,qi,:] @ Wv[o,:] + bv[o]
// --------------------------------------------------------------------------
__global__ void k_vproj(const _Float16* __restrict__ ctxpre_h, const _Float16* __restrict__ Wvh,
                        const float* __restrict__ bv, _Float16* __restrict__ ctx_h) {
  const int wave = threadIdx.x >> 5, lane = threadIdx.x & 31;
  const int half = lane >> 4, n = lane & 15;
  const int b = blockIdx.y;
  const int nt0 = (blockIdx.x * 8 + wave) * 4;       // 4 tiles, one head
  const int h = (nt0 * 16) >> 8;
  const _Float16* A = ctxpre_h + (((long)b * NH_ + h) * NC_) * H_;
  v8f acc[4] = {};
  for (int kt = 0; kt < H_ / 32; ++kt) {
    v16h a = load_a_frag_h(A + kt * 32, H_, lane);
#pragma unroll
    for (int t = 0; t < 4; ++t) {
      v16h bf = load_bT_frag_h(Wvh + (long)((nt0 + t) * 16) * H_ + kt * 32, H_, lane);
      acc[t] = wmma16x16x32(a, bf, acc[t]);
    }
  }
#pragma unroll
  for (int t = 0; t < 4; ++t) {
    const int o = (nt0 + t) * 16 + n;
    const float bias = bv[o];
#pragma unroll
    for (int r = 0; r < 8; ++r) {
      const int qi = r + 8 * half;
      ctx_h[((long)b * NC_ + qi) * H_ + o] = (_Float16)(acc[t][r] + bias);
    }
  }
}

// --------------------------------------------------------------------------
// Kernel 8: out_pre[b,qi,p] = ctx[b,qi,:] @ Wo[p,:] + bo[p]
// --------------------------------------------------------------------------
__global__ void k_outproj(const _Float16* __restrict__ ctx_h, const _Float16* __restrict__ Woh,
                          const float* __restrict__ bo, float* __restrict__ out_pre) {
  const int wave = threadIdx.x >> 5, lane = threadIdx.x & 31;
  const int half = lane >> 4, n = lane & 15;
  const int b = blockIdx.y;
  const int nt0 = (blockIdx.x * 8 + wave) * 4;
  const _Float16* A = ctx_h + (long)b * NC_ * H_;
  v8f acc[4] = {};
  for (int kt = 0; kt < H_ / 32; ++kt) {
    v16h a = load_a_frag_h(A + kt * 32, H_, lane);
#pragma unroll
    for (int t = 0; t < 4; ++t) {
      v16h bf = load_bT_frag_h(Woh + (long)((nt0 + t) * 16) * H_ + kt * 32, H_, lane);
      acc[t] = wmma16x16x32(a, bf, acc[t]);
    }
  }
#pragma unroll
  for (int t = 0; t < 4; ++t) {
    const int o = (nt0 + t) * 16 + n;
    const float bias = bo[o];
#pragma unroll
    for (int r = 0; r < 8; ++r) {
      const int qi = r + 8 * half;
      out_pre[((long)b * NC_ + qi) * H_ + o] = acc[t][r] + bias;
    }
  }
}

// --------------------------------------------------------------------------
// Kernel 9: RMSNorm rows of out_pre -> compressed
// --------------------------------------------------------------------------
__global__ void k_rmsnorm(const float* __restrict__ out_pre, const float* __restrict__ rms_w,
                          float* __restrict__ compressed) {
  __shared__ float red[256];
  const int tid = threadIdx.x;
  const float* p = out_pre + (long)blockIdx.x * H_;
  float ss = 0.0f;
  for (int i = tid; i < H_; i += 256) { const float v = p[i]; ss += v * v; }
  red[tid] = ss; __syncthreads();
  for (int s = 128; s > 0; s >>= 1) { if (tid < s) red[tid] += red[tid + s]; __syncthreads(); }
  const float inv = 1.0f / sqrtf(red[0] * (1.0f / H_) + 1e-6f);
  float* o = compressed + (long)blockIdx.x * H_;
  for (int i = tid; i < H_; i += 256) o[i] = p[i] * inv * rms_w[i];
}

// --------------------------------------------------------------------------
extern "C" void kernel_launch(void* const* d_in, const int* in_sizes, int n_in,
                              void* d_out, int out_size, void* d_ws, size_t ws_size,
                              hipStream_t stream) {
  const float* x   = (const float*)d_in[0];
  const float* W1  = (const float*)d_in[1];
  const float* b1  = (const float*)d_in[2];
  const float* W2  = (const float*)d_in[3];
  const float* b2  = (const float*)d_in[4];
  const float* cq  = (const float*)d_in[5];
  const float* Wq  = (const float*)d_in[6];
  const float* bq  = (const float*)d_in[7];
  const float* Wk  = (const float*)d_in[8];
  const float* bk  = (const float*)d_in[9];  (void)bk;   // constant in s -> cancels in softmax
  const float* Wv  = (const float*)d_in[10];
  const float* bv  = (const float*)d_in[11];
  const float* Wo  = (const float*)d_in[12];
  const float* bo  = (const float*)d_in[13];
  const float* rw  = (const float*)d_in[14];

  // d_out: compressed[B*NC*H] | importance[B*S] | attn_weights[B*NC*S]
  float* compressed = (float*)d_out;
  float* imp        = compressed + (long)B_ * NC_ * H_;
  float* attn_w     = imp + (long)B_ * S_;

  // workspace layout (all offsets 256B-aligned)
  char* w = (char*)d_ws;
  auto alloc = [&](size_t bytes) { char* p = w; w += (bytes + 255) & ~size_t(255); return p; };
  _Float16* xh      = (_Float16*)alloc(sizeof(_Float16) * (size_t)B_ * S_ * H_);   // 64 MB
  _Float16* W1h     = (_Float16*)alloc(sizeof(_Float16) * (size_t)(H_/4) * H_);
  _Float16* Wvh     = (_Float16*)alloc(sizeof(_Float16) * (size_t)H_ * H_);
  _Float16* Woh     = (_Float16*)alloc(sizeof(_Float16) * (size_t)H_ * H_);
  _Float16* qkh     = (_Float16*)alloc(sizeof(_Float16) * (size_t)NH_ * NC_ * H_);
  _Float16* attnh   = (_Float16*)alloc(sizeof(_Float16) * (size_t)B_ * NH_ * NC_ * S_);
  _Float16* ctxpre_h= (_Float16*)alloc(sizeof(_Float16) * (size_t)B_ * NH_ * NC_ * H_);
  _Float16* ctx_h   = (_Float16*)alloc(sizeof(_Float16) * (size_t)B_ * NC_ * H_);
  float*    q       = (float*)alloc(sizeof(float) * (size_t)NC_ * H_);
  float*    qk      = (float*)alloc(sizeof(float) * (size_t)NH_ * NC_ * H_);
  float*    attn    = (float*)alloc(sizeof(float) * (size_t)B_ * NH_ * NC_ * S_);  // 8 MB
  float*    out_pre = (float*)alloc(sizeof(float) * (size_t)B_ * NC_ * H_);

  k_cvt16<<<(int)(((size_t)B_ * S_ * H_) / 2048), 256, 0, stream>>>(x, xh);
  k_cvt16<<<(int)(((size_t)(H_/4) * H_) / 2048), 256, 0, stream>>>(W1, W1h);
  k_cvt16<<<(int)(((size_t)H_ * H_) / 2048), 256, 0, stream>>>(Wv, Wvh);
  k_cvt16<<<(int)(((size_t)H_ * H_) / 2048), 256, 0, stream>>>(Wo, Woh);

  k_importance<<<(B_ * S_) / 128, 256, 0, stream>>>(xh, W1h, b1, W2, b2, imp);
  k_qproj<<<(NC_ * H_) / 256, 256, 0, stream>>>(cq, Wq, bq, q);
  k_qkfold<<<(NH_ * NC_ * H_) / 256, 256, 0, stream>>>(q, Wk, qk);
  k_cvt16<<<(int)(((size_t)NH_ * NC_ * H_) / 2048), 256, 0, stream>>>(qk, qkh);

  k_scores<<<dim3(S_ / 128, B_), 256, 0, stream>>>(xh, qkh, imp, attn);
  k_softmax<<<B_ * NH_ * NC_, 256, 0, stream>>>(attn, attnh);
  k_attnmean<<<(B_ * NC_ * S_) / 256, 256, 0, stream>>>(attn, attn_w);

  k_ctxpre<<<dim3(H_ / 128, B_), 256, 0, stream>>>(attnh, xh, ctxpre_h);
  k_vproj<<<dim3(2, B_), 256, 0, stream>>>(ctxpre_h, Wvh, bv, ctx_h);
  k_outproj<<<dim3(2, B_), 256, 0, stream>>>(ctx_h, Woh, bo, out_pre);
  k_rmsnorm<<<B_ * NC_, 256, 0, stream>>>(out_pre, rw, compressed);
}
